// KascadeAnchorAttention_53386443489642
// MI455X (gfx1250) — matrix operand order repaired and checked by reading, not compiled, not run on hardware
//
#include <hip/hip_runtime.h>
#include <hip/hip_bf16.h>

typedef __attribute__((ext_vector_type(16))) _Float16 v16h;
typedef __attribute__((ext_vector_type(8)))  _Float16 v8h;
typedef __attribute__((ext_vector_type(8)))  float    v8f;
typedef __attribute__((ext_vector_type(4)))  float    f4;
typedef __attribute__((ext_vector_type(2)))  __fp16   half2v;
typedef __attribute__((ext_vector_type(4)))  __fp16   half4v;

#define B_  2
#define S_  2048
#define H_  16
#define DH_ 128
#define D_  2048          // H_*DH_
#define NT_ 16            // num tiles (S_/128)
#define TOPK_ 8
#define SCALE_ 0.08838834764831845f   // 1/sqrt(128)

__device__ __forceinline__ v8f wmma16(v16h a, v16h b, v8f c) {
  return __builtin_amdgcn_wmma_f32_16x16x32_f16(
      false, a, false, b, (short)0, c, false, false);
}

__device__ __forceinline__ v16h cat16(v8h lo, v8h hi) {
  return __builtin_shufflevector(lo, hi, 0,1,2,3,4,5,6,7,8,9,10,11,12,13,14,15);
}
// A-fragment: lane holds row[base..base+7] and row[base+16..base+23]
__device__ __forceinline__ v16h ldsA(const _Float16* p) {
  return cat16(*(const v8h*)p, *(const v8h*)(p + 16));
}
// B-fragment (operand stored N-major): lane holds k[base..base+7] and k[base+16..base+23]
__device__ __forceinline__ v16h ldsB(const _Float16* p) {
  return cat16(*(const v8h*)p, *(const v8h*)(p + 16));
}
// 4 x f32 -> 4 x f16 via v_cvt_pk_rtz_f16_f32 (2 converts total)
__device__ __forceinline__ half4v pack4(f4 x) {
  half2v lo = __builtin_amdgcn_cvt_pkrtz(x.x, x.y);
  half2v hi = __builtin_amdgcn_cvt_pkrtz(x.z, x.w);
  return __builtin_shufflevector(lo, hi, 0, 1, 2, 3);
}

// DPP16 row_xmask cross-lane moves: VALU-only reductions over 16-lane groups
// (replaces ds_bpermute-based __shfl_xor; no LDS traffic).
template <int CTRL>
__device__ __forceinline__ float dppmovf(float x) {
  return __int_as_float(
      __builtin_amdgcn_update_dpp(0, __float_as_int(x), CTRL, 0xf, 0xf, true));
}
__device__ __forceinline__ float rowmax16(float x) {
  x = fmaxf(x, dppmovf<0x168>(x));   // row_xmask:8
  x = fmaxf(x, dppmovf<0x164>(x));   // row_xmask:4
  x = fmaxf(x, dppmovf<0x162>(x));   // row_xmask:2
  x = fmaxf(x, dppmovf<0x161>(x));   // row_xmask:1
  return x;
}
__device__ __forceinline__ float rowsum16(float x) {
  x += dppmovf<0x168>(x);
  x += dppmovf<0x164>(x);
  x += dppmovf<0x162>(x);
  x += dppmovf<0x161>(x);
  return x;
}

// ---------------------------------------------------------------------------
// fp32-in / fp32-out GEMM via f16 WMMA.  C[M,N] = A[M,K] * B[K,N].
// Compile-time dims; block tile 128x128, 256 threads (8 waves), K-step 32.
// Wave w computes rows w*16..w*16+15 across all 128 cols: 8 WMMAs / K-step.
// ---------------------------------------------------------------------------
template <int M, int N, int K>
__global__ __launch_bounds__(256) void gemm_wmma_f16(
    const float* __restrict__ A, const float* __restrict__ Bm,
    float* __restrict__ C)
{
  __shared__ __align__(16) _Float16 As[128][40];   // [m][k], 80B pitch
  __shared__ __align__(16) _Float16 BsT[128][40];  // [n][k], 80B pitch

  const int tid  = threadIdx.x;
  const int lane = tid & 31;
  const int w    = tid >> 5;        // wave row strip 0..7
  const int bm = blockIdx.y * 128;
  const int bn = blockIdx.x * 128;

  const int m16 = lane & 15;
  const int kbA = (lane < 16) ? 0 : 8;
  const int kbB = (lane < 16) ? 0 : 8;
  const int nln = lane & 15;
  const int mb  = (lane < 16) ? 0 : 8;

  // static staging maps
  const int cA = (tid & 7) * 4;     // A: col group
  const int rA = tid >> 3;          // A: base row 0..31 (+32*j)
  const int cB = tid & 127;         // B: column (N dim)
  const int rgB = tid >> 7;         // B: row-group phase 0..1

  v8f acc[8];
  #pragma unroll
  for (int nf = 0; nf < 8; ++nf) { v8f z = {}; acc[nf] = z; }

  for (int k0 = 0; k0 < K; k0 += 32) {
    // ---- stage A tile 128x32 (row-major) ----
    #pragma unroll
    for (int j = 0; j < 4; ++j) {
      int r = rA + j * 32;
      f4 x = *(const f4*)&A[(size_t)(bm + r) * K + k0 + cA];
      *(half4v*)&As[r][cA] = pack4(x);
    }
    // ---- stage B tile 32x128 transposed (N-major) ----
    #pragma unroll
    for (int j = 0; j < 4; ++j) {
      int r0 = (rgB + 2 * j) * 4;   // 0..28 step 4
      f4 x;
      x.x = Bm[(size_t)(k0 + r0 + 0) * N + bn + cB];
      x.y = Bm[(size_t)(k0 + r0 + 1) * N + bn + cB];
      x.z = Bm[(size_t)(k0 + r0 + 2) * N + bn + cB];
      x.w = Bm[(size_t)(k0 + r0 + 3) * N + bn + cB];
      *(half4v*)&BsT[cB][r0] = pack4(x);
    }
    __syncthreads();

    v16h af = ldsA(&As[w * 16 + m16][kbA]);
    #pragma unroll
    for (int nf = 0; nf < 8; ++nf) {
      v16h bf = ldsB(&BsT[nf * 16 + nln][kbB]);
      acc[nf] = wmma16(af, bf, acc[nf]);
    }
    __syncthreads();
  }

  const int row0 = bm + w * 16;
  #pragma unroll
  for (int nf = 0; nf < 8; ++nf) {
    #pragma unroll
    for (int v = 0; v < 8; ++v)
      C[(size_t)(row0 + mb + v) * N + bn + nf * 16 + nln] = acc[nf][v];
  }
}

// ---------------------------------------------------------------------------
// RoPE (rotate_half form) applied in place to Q and K.
// ---------------------------------------------------------------------------
__global__ __launch_bounds__(256) void rope_kernel(
    float* __restrict__ q, float* __restrict__ k,
    const float* __restrict__ cosT, const float* __restrict__ sinT)
{
  size_t idx = (size_t)blockIdx.x * 256 + threadIdx.x;   // B*S*H*64 total
  int d = idx & 63;
  int h = (int)((idx >> 6) & 15);
  int s = (int)((idx >> 10) & 2047);
  int b = (int)(idx >> 21);
  float c  = cosT[s * 64 + d];
  float sn = sinT[s * 64 + d];
  size_t base = ((size_t)(b * S_ + s)) * D_ + h * DH_;
  float q0 = q[base + d], q1 = q[base + 64 + d];
  q[base + d]      = q0 * c - q1 * sn;
  q[base + 64 + d] = q1 * c + q0 * sn;
  float k0 = k[base + d], k1 = k[base + 64 + d];
  k[base + d]      = k0 * c - k1 * sn;
  k[base + 64 + d] = k1 * c + k0 * sn;
}

// ---------------------------------------------------------------------------
// Flash-style causal attention.  grid = (S/128, B*H), 256 threads (8 waves).
// K chunk row-major (dh contiguous -> QK^T B-frags contiguous);
// V chunk dh-major (key contiguous -> PV B-frags contiguous).
// ---------------------------------------------------------------------------
__global__ __launch_bounds__(256) void attn_kernel(
    const float* __restrict__ Q, const float* __restrict__ Kv,
    const float* __restrict__ V, float* __restrict__ O)
{
  __shared__ __align__(16) _Float16 Ks[64][136];    // [key][dh], 272B pitch
  __shared__ __align__(16) _Float16 VsT[128][72];   // [dh][key], 144B pitch
  __shared__ __align__(16) _Float16 Ps[8][16 * 32]; // per-wave P scratch

  const int bh = blockIdx.y;
  const int b  = bh >> 4, h = bh & 15;
  const int qb = blockIdx.x;              // query block of 128
  const int tid  = threadIdx.x;
  const int lane = tid & 31;
  const int w    = tid >> 5;
  const int qrow0 = qb * 128 + w * 16;

  const int m16 = lane & 15;
  const int kbA = (lane < 16) ? 0 : 8;
  const int kbB = (lane < 16) ? 0 : 8;
  const int nln = lane & 15;
  const int mb  = (lane < 16) ? 0 : 8;

  // static staging maps
  const int cK = (tid & 31) * 4;    // K: dh group
  const int rK = tid >> 5;          // K: base key row 0..7 (+8*j)
  const int cV = tid & 127;         // V: dh index
  const int rgV = tid >> 7;         // V: key-group phase 0..1

  // Q strip -> 4 A-fragments (dh = 128 split into 4x32), straight from global
  v16h qf[4];
  const float* qbase = Q + ((size_t)(b * S_) + qrow0) * D_ + h * DH_;
  #pragma unroll
  for (int c = 0; c < 4; ++c) {
    #pragma unroll
    for (int i = 0; i < 8; ++i) {
      qf[c][i]     = (_Float16)qbase[(size_t)m16 * D_ + c * 32 + kbA + i];
      qf[c][8 + i] = (_Float16)qbase[(size_t)m16 * D_ + c * 32 + kbA + 16 + i];
    }
  }

  v8f o[8];
  #pragma unroll
  for (int d = 0; d < 8; ++d) { v8f z = {}; o[d] = z; }
  float mrow[8], lrow[8];
  #pragma unroll
  for (int v = 0; v < 8; ++v) { mrow[v] = -3.0e38f; lrow[v] = 0.f; }

  const int nChunks = (qb + 1) * 2;       // keys < (qb+1)*128
  for (int kc = 0; kc < nChunks; ++kc) {
    const int key0 = kc * 64;
    // ---- stage K chunk 64x128 row-major ----
    #pragma unroll
    for (int j = 0; j < 8; ++j) {
      int r = rK + j * 8;
      f4 x = *(const f4*)&Kv[((size_t)(b * S_) + key0 + r) * D_ + h * DH_ + cK];
      *(half4v*)&Ks[r][cK] = pack4(x);
    }
    // ---- stage V chunk 64x128 transposed (dh-major) ----
    #pragma unroll
    for (int j = 0; j < 8; ++j) {
      int r0 = (rgV + 2 * j) * 4;   // 0..60 step 4
      f4 x;
      x.x = V[((size_t)(b * S_) + key0 + r0 + 0) * D_ + h * DH_ + cV];
      x.y = V[((size_t)(b * S_) + key0 + r0 + 1) * D_ + h * DH_ + cV];
      x.z = V[((size_t)(b * S_) + key0 + r0 + 2) * D_ + h * DH_ + cV];
      x.w = V[((size_t)(b * S_) + key0 + r0 + 3) * D_ + h * DH_ + cV];
      *(half4v*)&VsT[cV][r0] = pack4(x);
    }
    __syncthreads();

    #pragma unroll
    for (int kk = 0; kk < 2; ++kk) {      // 32-key sub-steps
      // ---- S = Q @ K^T (two 16x16 frags = 32 keys) ----
      v8f s0 = {}; v8f s1 = {};
      #pragma unroll
      for (int c = 0; c < 4; ++c) {
        v16h b0 = ldsB(&Ks[kk * 32 + nln][c * 32 + kbB]);
        v16h b1 = ldsB(&Ks[kk * 32 + 16 + nln][c * 32 + kbB]);
        s0 = wmma16(qf[c], b0, s0);
        s1 = wmma16(qf[c], b1, s1);
      }
      // ---- mask + online softmax (DPP row reductions, no LDS) ----
      const int k0g = key0 + kk * 32;
      #pragma unroll
      for (int v = 0; v < 8; ++v) {
        int qi = qrow0 + mb + v;
        float x0 = s0[v] * SCALE_;
        float x1 = s1[v] * SCALE_;
        if (k0g + nln > qi)      x0 = -1.0e30f;
        if (k0g + 16 + nln > qi) x1 = -1.0e30f;
        float mx = rowmax16(fmaxf(x0, x1));
        float mnew = fmaxf(mrow[v], mx);
        float corr = __expf(mrow[v] - mnew);
        float p0 = __expf(x0 - mnew);
        float p1 = __expf(x1 - mnew);
        float rs = rowsum16(p0 + p1);
        lrow[v] = lrow[v] * corr + rs;
        mrow[v] = mnew;
        #pragma unroll
        for (int d = 0; d < 8; ++d) o[d][v] *= corr;
        Ps[w][(mb + v) * 32 + nln]      = (_Float16)p0;
        Ps[w][(mb + v) * 32 + 16 + nln] = (_Float16)p1;
      }
      // wave-private LDS round trip: wait for our own ds stores
      asm volatile("s_wait_dscnt 0" ::: "memory");
      // ---- repack P as A-fragment (16x32) ----
      v16h pa = ldsA(&Ps[w][m16 * 32 + kbA]);
      // ---- O += P @ V (8 N-subtiles of 16) ----
      #pragma unroll
      for (int df = 0; df < 8; ++df) {
        v16h vb = ldsB(&VsT[df * 16 + nln][kk * 32 + kbB]);
        o[df] = wmma16(pa, vb, o[df]);
      }
    }
    __syncthreads();
  }

  // epilogue: normalize and store
  #pragma unroll
  for (int v = 0; v < 8; ++v) {
    float inv = 1.0f / lrow[v];
    int qi = qrow0 + mb + v;
    float* ob = O + ((size_t)(b * S_) + qi) * D_ + h * DH_;
    #pragma unroll
    for (int df = 0; df < 8; ++df)
      ob[df * 16 + nln] = o[df][v] * inv;
  }
}

// ---------------------------------------------------------------------------
// Tile scoring: softmax over representative rows, per-tile max, stable top-8.
// grid = B*H blocks, 256 threads.
// ---------------------------------------------------------------------------
__global__ __launch_bounds__(256) void tilescore_kernel(
    const float* __restrict__ Q, const float* __restrict__ Kv,
    int* __restrict__ idxOut)
{
  __shared__ __align__(16) float qrep[16][128];
  __shared__ float red[256];
  __shared__ unsigned int tsc[16][16];

  const int bh = blockIdx.x;
  const int b = bh >> 4, h = bh & 15;
  const int tid = threadIdx.x;

  for (int i = tid; i < 16 * 128; i += 256) {
    int t = i >> 7, d = i & 127;
    int rep = t * 128 + 127;
    qrep[t][d] = Q[((size_t)(b * S_) + rep) * D_ + h * DH_ + d];
  }
  tsc[tid >> 4][tid & 15] = 0u;
  __syncthreads();

  for (int t = 0; t < 16; ++t) {
    const int rep = t * 128 + 127;
    float lg[8];
    float lmax = -3.0e38f;
    #pragma unroll
    for (int i = 0; i < 8; ++i) {
      int k = tid + i * 256;
      const f4* kr4 = (const f4*)(Kv + ((size_t)(b * S_) + k) * D_ + h * DH_);
      const f4* q4  = (const f4*)&qrep[t][0];
      float acc = 0.f;
      #pragma unroll 8
      for (int d = 0; d < 32; ++d) {
        f4 a = q4[d], kk = kr4[d];
        acc += a.x * kk.x + a.y * kk.y + a.z * kk.z + a.w * kk.w;
      }
      acc *= SCALE_;
      if (k > rep) acc = -1.0e10f;
      lg[i] = acc;
      lmax = fmaxf(lmax, acc);
    }
    red[tid] = lmax; __syncthreads();
    for (int s = 128; s > 0; s >>= 1) {
      if (tid < s) red[tid] = fmaxf(red[tid], red[tid + s]);
      __syncthreads();
    }
    float m = red[0]; __syncthreads();
    float ls = 0.f;
    #pragma unroll
    for (int i = 0; i < 8; ++i) ls += __expf(lg[i] - m);
    red[tid] = ls; __syncthreads();
    for (int s = 128; s > 0; s >>= 1) {
      if (tid < s) red[tid] += red[tid + s];
      __syncthreads();
    }
    float inv = 1.0f / red[0]; __syncthreads();
    #pragma unroll
    for (int i = 0; i < 8; ++i) {
      int k = tid + i * 256;
      float wv = __expf(lg[i] - m) * inv;      // >= 0 -> uint order == float order
      atomicMax(&tsc[t][k >> 7], __float_as_uint(wv));
    }
    __syncthreads();
  }

  if (tid < 16) {
    float sc[16]; bool used[16];
    for (int j = 0; j < 16; ++j) { sc[j] = __uint_as_float(tsc[tid][j]); used[j] = false; }
    int* op = idxOut + ((size_t)bh * 16 + tid) * TOPK_;
    for (int i = 0; i < TOPK_; ++i) {
      int best = 0; float bv = -1.f;
      for (int j = 0; j < 16; ++j)
        if (!used[j] && sc[j] > bv) { bv = sc[j]; best = j; }
      used[best] = true;
      op[i] = best;
    }
  }
}

// ---------------------------------------------------------------------------
extern "C" void kernel_launch(void* const* d_in, const int* in_sizes, int n_in,
                              void* d_out, int out_size, void* d_ws, size_t ws_size,
                              hipStream_t stream) {
  const float* x    = (const float*)d_in[0];
  const float* wq   = (const float*)d_in[1];
  const float* wk   = (const float*)d_in[2];
  const float* wv   = (const float*)d_in[3];
  const float* wo   = (const float*)d_in[4];
  const float* cosT = (const float*)d_in[5];
  const float* sinT = (const float*)d_in[6];

  const size_t elems = (size_t)B_ * S_ * D_;   // 8,388,608
  float* qb = (float*)d_ws;
  float* kb = qb + elems;
  float* vb = kb + elems;
  float* ab = vb + elems;

  float* outF  = (float*)d_out;
  int*   idxO  = (int*)(outF + elems);

  constexpr int M = B_ * S_, N = D_, K = D_;
  dim3 ggrid(N / 128, M / 128);   // 16 x 32

  gemm_wmma_f16<M, N, K><<<ggrid, 256, 0, stream>>>(x, wq, qb);
  gemm_wmma_f16<M, N, K><<<ggrid, 256, 0, stream>>>(x, wk, kb);
  gemm_wmma_f16<M, N, K><<<ggrid, 256, 0, stream>>>(x, wv, vb);

  const size_t pairs = (size_t)B_ * S_ * H_ * 64;   // 4,194,304
  rope_kernel<<<(unsigned)(pairs / 256), 256, 0, stream>>>(qb, kb, cosT, sinT);

  attn_kernel<<<dim3(S_ / 128, B_ * H_), 256, 0, stream>>>(qb, kb, vb, ab);

  tilescore_kernel<<<B_ * H_, 256, 0, stream>>>(qb, kb, idxO);

  gemm_wmma_f16<M, N, K><<<ggrid, 256, 0, stream>>>(ab, wo, outF);
}
